// Attention_36867999269267
// MI455X (gfx1250) — compile-verified, matrix operand order checked by reading, not run on hardware
//
#include <hip/hip_runtime.h>
#include <hip/hip_bf16.h>

// ---------------------------------------------------------------------------
// Attention block for MI455X (gfx1250, wave32, WMMA).
// Pipeline: cvt -> fused QKV WMMA GEMM -> RoPE/split -> flash attention (WMMA
// for QK^T and PV with online softmax) -> output-projection WMMA GEMM.
// All matmuls use v_wmma_f32_16x16x32_bf16 with f32 accumulation.
// Global->LDS staging uses CDNA5 async copies (global_load_async_to_lds_b128,
// ASYNCcnt) with double-buffered LDS tiles to overlap HBM traffic with WMMA.
// Workspace requirement: ~130 MB.
// ---------------------------------------------------------------------------

typedef __attribute__((ext_vector_type(16))) __bf16 v16bf;
typedef __attribute__((ext_vector_type(8)))  __bf16 v8bf;
typedef __attribute__((ext_vector_type(8)))  float  v8f;

#define DIM       2048
#define NHEADS    16
#define NKVHEADS  4
#define HEADDIM   128
#define BATCH     2
#define SEQ       2048
#define MROWS     (BATCH * SEQ)      // 4096
#define NQKV      3072               // 2048 Q + 512 K + 512 V

__device__ __forceinline__ v8f wmma_bf16(v16bf a, v16bf b, v8f c) {
  // (neg_a, A, neg_b, B, c_mod, C, reuse_a, reuse_b)
  return __builtin_amdgcn_wmma_f32_16x16x32_bf16(false, a, false, b, (short)0, c,
                                                 false, false);
}

// Build a 16x32 bf16 A-style fragment: elements 0..7 from `lo`, 8..15 from `hi`
// (ISA layout: lanes 0-15 hold K 0-7 / 16-23, lanes 16-31 hold K 8-15 / 24-31,
// handled by the caller's base-address math). Both pointers 16B-aligned.
__device__ __forceinline__ v16bf load_frag(const __bf16* lo, const __bf16* hi) {
  v8bf l = *(const v8bf*)lo;
  v8bf h = *(const v8bf*)hi;
  return __builtin_shufflevector(l, h, 0, 1, 2, 3, 4, 5, 6, 7,
                                        8, 9, 10, 11, 12, 13, 14, 15);
}

// CDNA5 async global->LDS copy, 16 bytes per lane. The LDS operand is the
// wave-relative LDS byte offset = low 32 bits of the generic shared pointer.
// Tracked by ASYNCcnt (s_wait_asynccnt), no VGPR round-trip.
__device__ __forceinline__ void async_copy16(void* ldsDst, const void* gsrc) {
  unsigned lds = (unsigned)(unsigned long long)ldsDst;
  asm volatile("global_load_async_to_lds_b128 %0, %1, off"
               :: "v"(lds), "v"(gsrc)
               : "memory");
}
__device__ __forceinline__ void wait_async0() {
  asm volatile("s_wait_asynccnt 0x0" ::: "memory");
}

// ---------------------------------------------------------------------------
// Conversion kernels
// ---------------------------------------------------------------------------
__global__ void cvt_f32_bf16(const float* __restrict__ src,
                             __bf16* __restrict__ dst, int n) {
  int i = blockIdx.x * blockDim.x + threadIdx.x;
  if (i < n) dst[i] = (__bf16)src[i];
}

// src [K][N] f32 (row-major)  ->  dst [N][K] bf16 (transposed)
__global__ void cvt_transpose_bf16(const float* __restrict__ src,
                                   __bf16* __restrict__ dst, int K, int N) {
  int i = blockIdx.x * blockDim.x + threadIdx.x;
  if (i >= N * K) return;
  int n = i / K;
  int k = i - n * K;
  dst[(size_t)n * K + k] = (__bf16)src[(size_t)k * N + n];
}

// ---------------------------------------------------------------------------
// Tiled bf16 WMMA GEMM:  C[M][N] (f32) = A[M][K] (bf16) * BT[N][K] (bf16)^T
// Block: 256 threads (8 waves), 128x128 output tile, K stepped by 32.
// Wave grid 4x2: each wave owns 32(M) x 64(N) = 2x4 16x16 C fragments.
// Double-buffered LDS tiles filled with async global->LDS copies.
// ---------------------------------------------------------------------------
__global__ __launch_bounds__(256) void gemm_bf16(const __bf16* __restrict__ A,
                                                 const __bf16* __restrict__ BT,
                                                 float* __restrict__ C,
                                                 int M, int N, int K) {
  __shared__ __bf16 As[2][128 * 32];
  __shared__ __bf16 Bs[2][128 * 32];

  const int tid   = threadIdx.x;
  const int lane  = tid & 31;
  const int wave  = tid >> 5;
  const int waveM = wave >> 1;        // 0..3
  const int waveN = wave & 1;         // 0..1
  const int mBase = blockIdx.y * 128;
  const int nBase = blockIdx.x * 128;

  const int rA  = lane & 15;          // fragment row within 16
  const int khA = (lane >> 4) << 3;   // 0 or 8 (K sub-offset)

  const int ldRow = tid >> 1;         // 0..127 (tile row this thread loads)
  const int ldOff = (tid & 1) * 16;   // 0 or 16 (K offset, elements)

  const __bf16* aRow = A + (size_t)(mBase + ldRow) * K + ldOff;
  const __bf16* bRow = BT + (size_t)(nBase + ldRow) * K + ldOff;

  auto issue_tile = [&](int buf, int k0) {
    async_copy16(&As[buf][ldRow * 32 + ldOff],     aRow + k0);
    async_copy16(&As[buf][ldRow * 32 + ldOff + 8], aRow + k0 + 8);
    async_copy16(&Bs[buf][ldRow * 32 + ldOff],     bRow + k0);
    async_copy16(&Bs[buf][ldRow * 32 + ldOff + 8], bRow + k0 + 8);
  };

  v8f acc[2][4] = {};

  issue_tile(0, 0);
  for (int k0 = 0; k0 < K; k0 += 32) {
    const int buf = (k0 >> 5) & 1;
    wait_async0();        // this wave's copies for tile `buf` have landed
    __syncthreads();      // ... and so have every other wave's
    if (k0 + 32 < K) issue_tile(buf ^ 1, k0 + 32);   // overlap next tile
    if (k0 + 64 < K) {
      __builtin_prefetch(aRow + k0 + 64, 0, 1);      // prime GL2 further out
      __builtin_prefetch(bRow + k0 + 64, 0, 1);
    }

    v16bf af[2], bf[4];
#pragma unroll
    for (int mf = 0; mf < 2; ++mf) {
      const __bf16* p = &As[buf][(waveM * 32 + mf * 16 + rA) * 32 + khA];
      af[mf] = load_frag(p, p + 16);
    }
#pragma unroll
    for (int nf = 0; nf < 4; ++nf) {
      const __bf16* p = &Bs[buf][(waveN * 64 + nf * 16 + rA) * 32 + khA];
      bf[nf] = load_frag(p, p + 16);
    }
#pragma unroll
    for (int mf = 0; mf < 2; ++mf)
#pragma unroll
      for (int nf = 0; nf < 4; ++nf)
        acc[mf][nf] = wmma_bf16(af[mf], bf[nf], acc[mf][nf]);
  }

  // C layout: element i -> row = i + 8*(lane>>4), col = lane & 15
  const int cg = (lane >> 4) << 3;
  const int cn = lane & 15;
#pragma unroll
  for (int mf = 0; mf < 2; ++mf)
#pragma unroll
    for (int nf = 0; nf < 4; ++nf)
#pragma unroll
      for (int i = 0; i < 8; ++i) {
        int r = mBase + waveM * 32 + mf * 16 + cg + i;
        int c = nBase + waveN * 64 + nf * 16 + cn;
        C[(size_t)r * N + c] = acc[mf][nf][i];
      }
}

// ---------------------------------------------------------------------------
// RoPE + head split. qkv f32 [B*S][3072] -> Qb [B][16][S][128] bf16 (scaled by
// 1/sqrt(128)), Kb [B][4][S][128] bf16, VTb [B][4][128][S] bf16 (transposed so
// PV B-fragments are contiguous loads).
// ---------------------------------------------------------------------------
__global__ __launch_bounds__(256) void rope_split(const float* __restrict__ qkv,
                                                  const float* __restrict__ fcos,
                                                  const float* __restrict__ fsin,
                                                  __bf16* __restrict__ Qb,
                                                  __bf16* __restrict__ Kb,
                                                  __bf16* __restrict__ VTb) {
  const int tok = blockIdx.x;          // 0..4095
  const int b = tok >> 11;
  const int s = tok & 2047;
  const int tid = threadIdx.x;
  const float* row = qkv + (size_t)tok * NQKV;
  const float* cr = fcos + s * 64;
  const float* sr = fsin + s * 64;
  const float qscale = 0.08838834764831845f;  // 1/sqrt(128)

  for (int t = tid; t < NHEADS * 64; t += 256) {       // Q pairs
    int h = t >> 6, p = t & 63;
    float r0 = row[h * 128 + 2 * p];
    float i0 = row[h * 128 + 2 * p + 1];
    float c = cr[p], sn = sr[p];
    size_t base = (((size_t)b * NHEADS + h) * SEQ + s) * HEADDIM;
    Qb[base + 2 * p]     = (__bf16)((r0 * c - i0 * sn) * qscale);
    Qb[base + 2 * p + 1] = (__bf16)((r0 * sn + i0 * c) * qscale);
  }
  for (int t = tid; t < NKVHEADS * 64; t += 256) {     // K pairs
    int h = t >> 6, p = t & 63;
    float r0 = row[2048 + h * 128 + 2 * p];
    float i0 = row[2048 + h * 128 + 2 * p + 1];
    float c = cr[p], sn = sr[p];
    size_t base = (((size_t)b * NKVHEADS + h) * SEQ + s) * HEADDIM;
    Kb[base + 2 * p]     = (__bf16)(r0 * c - i0 * sn);
    Kb[base + 2 * p + 1] = (__bf16)(r0 * sn + i0 * c);
  }
  for (int t = tid; t < NKVHEADS * HEADDIM; t += 256) { // V transpose
    int h = t >> 7, d = t & 127;
    VTb[(((size_t)b * NKVHEADS + h) * HEADDIM + d) * SEQ + s] =
        (__bf16)row[2560 + h * 128 + d];
  }
}

// ---------------------------------------------------------------------------
// Flash attention. Grid (S/128, H, B), 256 threads = 8 waves x 16 q-rows.
// K/V^T tiles of 32 positions double-buffered in LDS via async copies;
// online softmax; P converted C-layout -> A-layout via per-wave LDS staging.
// ---------------------------------------------------------------------------
__global__ __launch_bounds__(256) void flash_attn(const __bf16* __restrict__ Qb,
                                                  const __bf16* __restrict__ Kb,
                                                  const __bf16* __restrict__ VTb,
                                                  __bf16* __restrict__ Ob) {
  __shared__ __bf16 Kt[2][32 * HEADDIM];   // [kp][d]   2 x 8 KB
  __shared__ __bf16 Vt[2][HEADDIM * 32];   // [d][kp]   2 x 8 KB
  __shared__ __bf16 Pl[8][16 * 32];        // per-wave P staging, 8 KB

  const int tid  = threadIdx.x;
  const int lane = tid & 31;
  const int wave = tid >> 5;
  const int h = blockIdx.y;
  const int b = blockIdx.z;
  const int kvh = h >> 2;               // GQA: 16 heads -> 4 kv heads
  const int qBase = blockIdx.x * 128;
  const int qRow0 = qBase + wave * 16;

  const int rA  = lane & 15;
  const int khA = (lane >> 4) << 3;

  // Q fragments for this wave's 16 rows (entire head dim, 4 x K=32 chunks)
  const __bf16* Qp = Qb + (((size_t)b * NHEADS + h) * SEQ + qRow0) * HEADDIM;
  v16bf qf[4];
#pragma unroll
  for (int dc = 0; dc < 4; ++dc) {
    const __bf16* p = Qp + (size_t)rA * HEADDIM + dc * 32 + khA;
    qf[dc] = load_frag(p, p + 16);
  }

  v8f o[8] = {};
  float mrow[8], lrow[8];
#pragma unroll
  for (int i = 0; i < 8; ++i) { mrow[i] = -3.0e38f; lrow[i] = 0.0f; }

  const __bf16* Kp = Kb + ((size_t)b * NKVHEADS + kvh) * SEQ * HEADDIM;
  const __bf16* Vp = VTb + ((size_t)b * NKVHEADS + kvh) * HEADDIM * SEQ;

  // per-thread staging assignments for the cooperative tile fills
  const int kRow  = tid >> 3;          // 0..31 (K tile row)
  const int kdOff = (tid & 7) * 16;    // 0..112 (K tile d offset)
  const int vRow  = tid >> 1;          // 0..127 (V^T tile row)
  const int vcOff = (tid & 1) * 16;    // 0 or 16 (V^T kp offset)

  auto issue_kv = [&](int buf, int k0) {
    const __bf16* ks = Kp + (size_t)(k0 + kRow) * HEADDIM + kdOff;
    async_copy16(&Kt[buf][kRow * HEADDIM + kdOff],     ks);
    async_copy16(&Kt[buf][kRow * HEADDIM + kdOff + 8], ks + 8);
    const __bf16* vs = Vp + (size_t)vRow * SEQ + k0 + vcOff;
    async_copy16(&Vt[buf][vRow * 32 + vcOff],     vs);
    async_copy16(&Vt[buf][vRow * 32 + vcOff + 8], vs + 8);
  };

  const int kEnd = qBase + 128;         // causal upper bound for this block
  issue_kv(0, 0);
  for (int k0 = 0; k0 < kEnd; k0 += 32) {
    const int buf = (k0 >> 5) & 1;
    wait_async0();
    __syncthreads();
    if (k0 + 32 < kEnd) issue_kv(buf ^ 1, k0 + 32);   // overlap next K/V tile

    // Scores: S[16x32] = Q(16x128) . K^T, via 2x4 WMMAs
    v8f sc[2];
#pragma unroll
    for (int kf = 0; kf < 2; ++kf) {
      v8f s = {};
#pragma unroll
      for (int dc = 0; dc < 4; ++dc) {
        const __bf16* p = &Kt[buf][(kf * 16 + rA) * HEADDIM + dc * 32 + khA];
        v16bf kb = load_frag(p, p + 16);
        s = wmma_bf16(qf[dc], kb, s);
      }
      sc[kf] = s;
    }

    // Online softmax (C layout: row = i + 8*(lane>>4), col = lane&15)
    const int colB = k0 + (lane & 15);
    const int rowG = qRow0 + ((lane >> 4) << 3);
    float alphaI[8];
#pragma unroll
    for (int i = 0; i < 8; ++i) {
      float c0 = (colB      <= rowG + i) ? sc[0][i] : -3.0e38f;
      float c1 = (colB + 16 <= rowG + i) ? sc[1][i] : -3.0e38f;
      float mx = fmaxf(c0, c1);
#pragma unroll
      for (int d = 1; d < 16; d <<= 1) mx = fmaxf(mx, __shfl_xor(mx, d, 32));
      float mnew = fmaxf(mrow[i], mx);
      float alpha = __expf(mrow[i] - mnew);
      float p0 = __expf(c0 - mnew);
      float p1 = __expf(c1 - mnew);
      float rs = p0 + p1;
#pragma unroll
      for (int d = 1; d < 16; d <<= 1) rs += __shfl_xor(rs, d, 32);
      lrow[i] = lrow[i] * alpha + rs;
      mrow[i] = mnew;
      alphaI[i] = alpha;
      // Stage P (bf16) into per-wave LDS in matrix row-major order
      __bf16* pl = &Pl[wave][(((lane >> 4) << 3) + i) * 32];
      pl[lane & 15]        = (__bf16)p0;
      pl[16 + (lane & 15)] = (__bf16)p1;
    }
#pragma unroll
    for (int oc = 0; oc < 8; ++oc)
#pragma unroll
      for (int i = 0; i < 8; ++i) o[oc][i] = o[oc][i] * alphaI[i];

    // LDS is in-order per wave; fence the compiler + ds counter anyway.
    asm volatile("s_wait_dscnt 0" ::: "memory");

    // Re-read P as an A-layout 16x32 fragment
    const __bf16* pb = &Pl[wave][rA * 32 + khA];
    v16bf pf = load_frag(pb, pb + 16);

    // O(16x128) += P(16x32) . V(32x128), 8 WMMAs over d chunks of 16
#pragma unroll
    for (int oc = 0; oc < 8; ++oc) {
      const __bf16* vb = &Vt[buf][(oc * 16 + rA) * 32 + khA];
      v16bf vf = load_frag(vb, vb + 16);
      o[oc] = wmma_bf16(pf, vf, o[oc]);
    }
  }

  // Normalize and write attn output as bf16, layout [B][S][H*D]
  float inv[8];
#pragma unroll
  for (int i = 0; i < 8; ++i) inv[i] = 1.0f / lrow[i];
  const int cg = (lane >> 4) << 3;
  const int cn = lane & 15;
#pragma unroll
  for (int oc = 0; oc < 8; ++oc)
#pragma unroll
    for (int i = 0; i < 8; ++i) {
      int t = qRow0 + cg + i;   // token within sequence
      size_t addr = ((size_t)b * SEQ + t) * (NHEADS * HEADDIM) +
                    h * HEADDIM + oc * 16 + cn;
      Ob[addr] = (__bf16)(o[oc][i] * inv[i]);
    }
}

// ---------------------------------------------------------------------------
// Host-side pipeline
// ---------------------------------------------------------------------------
extern "C" void kernel_launch(void* const* d_in, const int* in_sizes, int n_in,
                              void* d_out, int out_size, void* d_ws, size_t ws_size,
                              hipStream_t stream) {
  const float* x    = (const float*)d_in[0];
  const float* fcos = (const float*)d_in[1];
  const float* fsin = (const float*)d_in[2];
  const float* wq   = (const float*)d_in[3];
  const float* wk   = (const float*)d_in[4];
  const float* wv   = (const float*)d_in[5];
  const float* wo   = (const float*)d_in[6];
  float* out = (float*)d_out;

  char* ws = (char*)d_ws;
  size_t off = 0;
  auto alloc = [&](size_t bytes) -> void* {
    void* p = ws + off;
    off = (off + bytes + 255) & ~(size_t)255;
    return p;
  };
  __bf16* xb    = (__bf16*)alloc((size_t)MROWS * DIM * 2);        // 16 MB
  __bf16* wqkvT = (__bf16*)alloc((size_t)NQKV * DIM * 2);         // 12 MB
  __bf16* woT   = (__bf16*)alloc((size_t)DIM * DIM * 2);          //  8 MB
  float*  qkv   = (float*) alloc((size_t)MROWS * NQKV * 4);       // 48 MB
  __bf16* Qb    = (__bf16*)alloc((size_t)BATCH * NHEADS * SEQ * HEADDIM * 2);
  __bf16* Kb    = (__bf16*)alloc((size_t)BATCH * NKVHEADS * SEQ * HEADDIM * 2);
  __bf16* VTb   = (__bf16*)alloc((size_t)BATCH * NKVHEADS * HEADDIM * SEQ * 2);
  __bf16* attnO = (__bf16*)alloc((size_t)MROWS * DIM * 2);        // 16 MB

  // 1) precision conversion + weight transposes (one-time, bandwidth-trivial)
  cvt_f32_bf16<<<(MROWS * DIM) / 256, 256, 0, stream>>>(x, xb, MROWS * DIM);
  cvt_transpose_bf16<<<(2048 * 2048) / 256, 256, 0, stream>>>(wq, wqkvT, DIM, 2048);
  cvt_transpose_bf16<<<(512 * 2048) / 256, 256, 0, stream>>>(
      wk, wqkvT + (size_t)2048 * DIM, DIM, 512);
  cvt_transpose_bf16<<<(512 * 2048) / 256, 256, 0, stream>>>(
      wv, wqkvT + (size_t)2560 * DIM, DIM, 512);
  cvt_transpose_bf16<<<(2048 * 2048) / 256, 256, 0, stream>>>(wo, woT, DIM, DIM);

  // 2) fused QKV projection: [4096][2048] x [2048][3072] -> f32
  gemm_bf16<<<dim3(NQKV / 128, MROWS / 128), 256, 0, stream>>>(
      xb, wqkvT, qkv, MROWS, NQKV, DIM);

  // 3) RoPE + head split (+ V transpose, + fold 1/sqrt(d) into Q)
  rope_split<<<MROWS, 256, 0, stream>>>(qkv, fcos, fsin, Qb, Kb, VTb);

  // 4) causal flash attention
  flash_attn<<<dim3(SEQ / 128, NHEADS, BATCH), 256, 0, stream>>>(Qb, Kb, VTb, attnO);

  // 5) output projection -> d_out (f32)
  gemm_bf16<<<dim3(DIM / 128, MROWS / 128), 256, 0, stream>>>(
      attnO, woT, out, MROWS, DIM, DIM);
}
